// SwitchHeadCore_24146306138175
// MI455X (gfx1250) — compile-verified, matrix operand order checked by reading, not run on hardware
//
#include <hip/hip_runtime.h>
#include <hip/hip_bf16.h>
#include <cmath>

// ---------------------------------------------------------------------------
// SwitchHead MoE attention forward for MI455X (gfx1250).
// bf16 WMMA pipeline + Tensor Data Mover (TDM) tile staging.
// ---------------------------------------------------------------------------

typedef __bf16 bf16_t;
typedef __attribute__((ext_vector_type(16))) __bf16    v16bf;
typedef __attribute__((ext_vector_type(8)))  float     v8f;
typedef __attribute__((ext_vector_type(4)))  unsigned  v4u;
typedef __attribute__((ext_vector_type(8)))  int       v8i;
typedef __attribute__((ext_vector_type(4)))  int       v4i;

static constexpr int BDIM  = 4;
static constexpr int SDIM  = 2048;
static constexpr int DDIM  = 1024;
static constexpr int HDIM  = 8;
static constexpr int EDIM  = 4;
static constexpr int DHDIM = 128;
static constexpr int KSEL  = 2;
static constexpr int BS    = BDIM * SDIM;          // 8192 tokens
static constexpr int HE    = HDIM * EDIM;          // 32
static constexpr int HDH   = HDIM * DHDIM;         // 1024
static constexpr int HEDH  = HE * DHDIM;           // 4096

#define DEV_INLINE __device__ __forceinline__

#if __has_builtin(__builtin_amdgcn_tensor_load_to_lds)
#define HAVE_TDM 1
#else
#define HAVE_TDM 0
#endif

DEV_INLINE v8f wmma_bf16(v16bf a, v16bf b, v8f c) {
  return __builtin_amdgcn_wmma_f32_16x16x32_bf16(
      /*neg_a=*/false, a, /*neg_b=*/false, b,
      /*c_mod=*/(short)0, c, /*reuse_a=*/false, /*reuse_b=*/false);
}

// Load a 16x32 bf16 fragment whose 32 K-elements are contiguous (16 dwords)
// starting at rowptr.  ISA layout: lanes 0-15 hold K=0..7 (v0..3) and
// K=16..23 (v4..7); lanes 16-31 hold K=8..15 and K=24..31.
DEV_INLINE v16bf load_frag(const unsigned* rowptr, int lh) {
  union { v16bf v; unsigned u[8]; } f;
#pragma unroll
  for (int j = 0; j < 4; ++j) {
    f.u[j]     = rowptr[4 * lh + j];
    f.u[4 + j] = rowptr[8 + 4 * lh + j];
  }
  return f.v;
}

DEV_INLINE void wait_lds() {
#if __has_builtin(__builtin_amdgcn_s_wait_dscnt)
  __builtin_amdgcn_s_wait_dscnt(0);
#else
  asm volatile("s_wait_dscnt 0" ::: "memory");
#endif
}

#if HAVE_TDM
// LDS byte offset of a __shared__ object (addrspacecast -> ptrtoint).
DEV_INLINE unsigned lds_offset(const void* p) {
  return (unsigned)(unsigned long long)
      (__attribute__((address_space(3))) const char*)p;
}

// Issue one TDM 2D tile load: global (row-major, rows of `stride0` bf16
// elements) -> LDS at byte offset lds_off, tile = tile_d0 x tile_d1 elements
// (tile_d0 = contiguous dim).  Per-wave scalar op; caller gates by wave and
// must s_wait_tensorcnt before consumers read the LDS tile.
DEV_INLINE void tdm_load_2d(unsigned lds_off, const void* gaddr,
                            unsigned tile_d0, unsigned tile_d1,
                            unsigned stride0) {
  const unsigned long long ga = (unsigned long long)gaddr;
  const unsigned big = 1u << 30;           // tensor dims: never clamp (in-bounds)
  v4u g0;
  g0[0] = 1u;                              // count=1, user mode, no gather
  g0[1] = lds_off;                         // lds_addr (bytes)
  g0[2] = (unsigned)(ga & 0xffffffffu);    // global_addr[31:0]
  g0[3] = (unsigned)((ga >> 32) & 0x01ffffffu) | (2u << 30);  // addr[56:32]|type=2
  v8i g1;
  g1[0] = (int)(1u << 16);                 // wg_mask=0, data_size=1 (2 bytes)
  g1[1] = (int)((big & 0xffffu) << 16);                    // tensor_dim0[15:0]
  g1[2] = (int)((big >> 16) | ((big & 0xffffu) << 16));    // d0 hi | d1 lo
  g1[3] = (int)((big >> 16) | (tile_d0 << 16));            // d1 hi | tile_dim0
  g1[4] = (int)(tile_d1 & 0xffffu);                        // tile_dim1 (tile_dim2=0)
  g1[5] = (int)stride0;                                    // dim0_stride[31:0]
  g1[6] = 0;                                               // stride hi | dim1_stride lo
  g1[7] = 0;
  v4i z4 = {0, 0, 0, 0};
#if __clang_major__ >= 23
  v8i z8 = {0, 0, 0, 0, 0, 0, 0, 0};
  __builtin_amdgcn_tensor_load_to_lds(g0, g1, z4, z4, z8, 0);
#else
  __builtin_amdgcn_tensor_load_to_lds(g0, g1, z4, z4, 0);
#endif
}

DEV_INLINE void wait_tensor() { __builtin_amdgcn_s_wait_tensorcnt(0); }
#endif  // HAVE_TDM

// ---------------------------------------------------------------------------
// Generic bf16 GEMM:  C[M,N] = alpha * A[M,K] @ B[K,N]   (A,B bf16 row-major)
// Block tile 128(M) x 64(N), K step 32; 8 waves, each wave = 16M x 64N.
// A tile staged by the Tensor Data Mover; B tile staged+transposed by VALU.
// ---------------------------------------------------------------------------
template <bool WRITE_BF16>
__global__ __launch_bounds__(256) void gemm_bf16(
    const bf16_t* __restrict__ A, const bf16_t* __restrict__ B,
    void* __restrict__ Cout, int M, int N, int Kd, float alpha) {
  __shared__ bf16_t As[128][32];   // row-major A tile
  __shared__ bf16_t Bt[64][32];    // B tile transposed: Bt[n][k]

  const int tid  = threadIdx.x;
  const int wv   = tid >> 5;
  const int lane = tid & 31;
  const int lr   = lane & 15;
  const int lh   = lane >> 4;
  const int m0   = blockIdx.y * 128;
  const int n0   = blockIdx.x * 64;

  v8f acc[4];
#pragma unroll
  for (int nf = 0; nf < 4; ++nf)
#pragma unroll
    for (int r = 0; r < 8; ++r) acc[nf][r] = 0.0f;

  const unsigned* AsU = (const unsigned*)&As[0][0];
  const unsigned* BtU = (const unsigned*)&Bt[0][0];
#if HAVE_TDM
  const unsigned As_off = lds_offset(&As[0][0]);
#endif

  for (int k0 = 0; k0 < Kd; k0 += 32) {
    __syncthreads();
#if HAVE_TDM
    if (wv == 0)  // one DMA for the whole 128x32 A tile
      tdm_load_2d(As_off, A + (size_t)m0 * Kd + k0, 32u, 128u, (unsigned)Kd);
#else
    {  // manual A stage: 128 x 32, each thread moves 16 bf16 (2 x uint4)
      const int row = tid >> 1, half = tid & 1;
      const unsigned* src = (const unsigned*)A +
          (size_t)(m0 + row) * (Kd >> 1) + (k0 >> 1) + half * 8;
      uint4 a0 = ((const uint4*)src)[0];
      uint4 a1 = ((const uint4*)src)[1];
      unsigned* dst = (unsigned*)&As[0][0] + row * 16 + half * 8;
      ((uint4*)dst)[0] = a0;
      ((uint4*)dst)[1] = a1;
    }
#endif
    {  // stage B transposed: 32 x 64 -> Bt[64][32]
      const int kk = tid >> 3, ng = (tid & 7) * 8;
      const bf16_t* src = B + (size_t)(k0 + kk) * N + n0 + ng;
      uint4 bv = *(const uint4*)src;
      const bf16_t* bp = (const bf16_t*)&bv;
#pragma unroll
      for (int i = 0; i < 8; ++i) Bt[ng + i][kk] = bp[i];
      if (k0 + 32 < Kd)
        __builtin_prefetch((const void*)(src + (size_t)32 * N), 0, 1);
    }
#if HAVE_TDM
    if (wv == 0) wait_tensor();  // TDM done before barrier releases consumers
#endif
    __syncthreads();

    const v16bf afrag = load_frag(AsU + (wv * 16 + lr) * 16, lh);
#pragma unroll
    for (int nf = 0; nf < 4; ++nf) {
      const v16bf bfrag = load_frag(BtU + (nf * 16 + lr) * 16, lh);
      acc[nf] = wmma_bf16(afrag, bfrag, acc[nf]);
    }
  }

#pragma unroll
  for (int nf = 0; nf < 4; ++nf)
#pragma unroll
    for (int r = 0; r < 8; ++r) {
      const int row = m0 + wv * 16 + r + 8 * lh;
      const int col = n0 + nf * 16 + lr;
      const float v = acc[nf][r] * alpha;
      if (WRITE_BF16)
        ((bf16_t*)Cout)[(size_t)row * N + col] = (bf16_t)v;
      else
        ((float*)Cout)[(size_t)row * N + col] = v;
    }
}

// ---------------------------------------------------------------------------
// Flash attention (causal).  Q,K: bf16 [BS, H*DH]; V: bf16 [B,H,S,DH].
// Out: f32 [B,S,H,DH].  Block = 128 threads (4 waves); block covers 64 q rows
// (16 per wave); KV streamed in tiles of 32 keys through LDS.
// K tile staged by TDM; V tile staged+transposed by VALU.
// ---------------------------------------------------------------------------
__global__ __launch_bounds__(128) void flash_attn(
    const bf16_t* __restrict__ Q, const bf16_t* __restrict__ Kmat,
    const bf16_t* __restrict__ V, float* __restrict__ Out) {
  __shared__ bf16_t Ks[32][128];      // K tile, row-major (key, dh)
  __shared__ bf16_t Vt[128][32];      // V tile transposed (dh, key)
  __shared__ bf16_t Pl[4][16][32];    // per-wave P tile (row, key)

  const int tid  = threadIdx.x;
  const int wv   = tid >> 5;
  const int lane = tid & 31;
  const int lr   = lane & 15;
  const int lh   = lane >> 4;
  const int qt   = blockIdx.x;               // q tile of 64 rows
  const int bh   = blockIdx.y;
  const int b    = bh / HDIM;
  const int h    = bh % HDIM;

  // Q fragments for this wave's 16 rows (4 K-chunks of 32 over DH=128)
  const int qrow0 = qt * 64 + wv * 16 + lr;
  const unsigned* qrp = (const unsigned*)Q +
      (size_t)(b * SDIM + qrow0) * (HDH / 2) + h * (DHDIM / 2);
  v16bf qf[4];
#pragma unroll
  for (int c = 0; c < 4; ++c) qf[c] = load_frag(qrp + 16 * c, lh);

  v8f oacc[8];
#pragma unroll
  for (int t = 0; t < 8; ++t)
#pragma unroll
    for (int r = 0; r < 8; ++r) oacc[t][r] = 0.0f;
  float mrow[8], lrow[8];
#pragma unroll
  for (int r = 0; r < 8; ++r) { mrow[r] = -1e30f; lrow[r] = 0.0f; }

#if HAVE_TDM
  const unsigned Ks_off = lds_offset(&Ks[0][0]);
#endif

  const int ntiles = 2 * (qt + 1);
  for (int t = 0; t < ntiles; ++t) {
    const int kb = t * 32;
    __syncthreads();
#if HAVE_TDM
    if (wv == 0)  // 32 keys x 128 dh, row stride H*DH elements
      tdm_load_2d(Ks_off,
                  Kmat + (size_t)(b * SDIM + kb) * HDH + h * DHDIM,
                  128u, 32u, (unsigned)HDH);
#else
    {  // manual K stage (row-major)
      const int key = tid & 31, qq = tid >> 5, dh0 = qq * 32;
      const uint4* ksrc = (const uint4*)(Kmat +
          (size_t)(b * SDIM + kb + key) * HDH + h * DHDIM + dh0);
      uint4 k0 = ksrc[0], k1 = ksrc[1], k2 = ksrc[2], k3 = ksrc[3];
      *(uint4*)&Ks[key][dh0 + 0]  = k0;
      *(uint4*)&Ks[key][dh0 + 8]  = k1;
      *(uint4*)&Ks[key][dh0 + 16] = k2;
      *(uint4*)&Ks[key][dh0 + 24] = k3;
    }
#endif
    {  // stage V tile transposed (dh, key)
      const int key = tid & 31, qq = tid >> 5, dh0 = qq * 32;
      const uint4* vsrc = (const uint4*)(V +
          (size_t)((b * HDIM + h) * SDIM + kb + key) * DHDIM + dh0);
      uint4 varr[4] = {vsrc[0], vsrc[1], vsrc[2], vsrc[3]};
      const bf16_t* vp = (const bf16_t*)varr;
#pragma unroll
      for (int i = 0; i < 32; ++i) Vt[dh0 + i][key] = vp[i];
    }
#if HAVE_TDM
    if (wv == 0) wait_tensor();
#endif
    __syncthreads();

    // scores: 16 q-rows x 32 keys, via 2 N-tiles x 4 K-chunks of WMMA
    v8f sc[2];
#pragma unroll
    for (int n2 = 0; n2 < 2; ++n2) {
      v8f a;
#pragma unroll
      for (int r = 0; r < 8; ++r) a[r] = 0.0f;
      const unsigned* krow = (const unsigned*)&Ks[n2 * 16 + lr][0];
#pragma unroll
      for (int c = 0; c < 4; ++c)
        a = wmma_bf16(qf[c], load_frag(krow + 16 * c, lh), a);
      sc[n2] = a;
    }

    // online softmax per row (row M spread across a 16-lane group)
    float p0s[8], p1s[8];
#pragma unroll
    for (int r = 0; r < 8; ++r) {
      const int qr = qt * 64 + wv * 16 + r + 8 * lh;
      float s0 = sc[0][r];
      float s1 = sc[1][r];
      if (kb + lr > qr)       s0 = -1e30f;
      if (kb + 16 + lr > qr)  s1 = -1e30f;
      float mx = fmaxf(s0, s1);
#pragma unroll
      for (int d = 8; d >= 1; d >>= 1) mx = fmaxf(mx, __shfl_xor(mx, d, 32));
      const float mnew  = fmaxf(mrow[r], mx);
      const float scale = __expf(mrow[r] - mnew);
      const float p0 = __expf(s0 - mnew);
      const float p1 = __expf(s1 - mnew);
      float rs = p0 + p1;
#pragma unroll
      for (int d = 8; d >= 1; d >>= 1) rs += __shfl_xor(rs, d, 32);
      lrow[r] = lrow[r] * scale + rs;
      mrow[r] = mnew;
#pragma unroll
      for (int tt = 0; tt < 8; ++tt) oacc[tt][r] *= scale;
      p0s[r] = p0;
      p1s[r] = p1;
    }

    // C-layout -> A-layout for P via per-wave LDS tile
#pragma unroll
    for (int r = 0; r < 8; ++r) {
      Pl[wv][r + 8 * lh][lr]      = (bf16_t)p0s[r];
      Pl[wv][r + 8 * lh][16 + lr] = (bf16_t)p1s[r];
    }
    wait_lds();

    const v16bf pf = load_frag((const unsigned*)&Pl[wv][lr][0], lh);
#pragma unroll
    for (int tt = 0; tt < 8; ++tt) {
      const v16bf vf = load_frag((const unsigned*)&Vt[tt * 16 + lr][0], lh);
      oacc[tt] = wmma_bf16(pf, vf, oacc[tt]);
    }
  }

  // epilogue: out[b, s, h, dh] = O / l
#pragma unroll
  for (int tt = 0; tt < 8; ++tt)
#pragma unroll
    for (int r = 0; r < 8; ++r) {
      const int qr = qt * 64 + wv * 16 + r + 8 * lh;
      Out[((size_t)(b * SDIM + qr) * HDIM + h) * DHDIM + tt * 16 + lr] =
          oacc[tt][r] / lrow[r];
    }
}

// ---------------------------------------------------------------------------
// Gating: per token compute 32 logits (H*E), sigmoid, top-2 of 4 per head.
// Writes dense gate [BS,H,E] (zeros at unselected) + indices as floats.
// ---------------------------------------------------------------------------
__global__ __launch_bounds__(256) void gate_kernel(
    const float* __restrict__ X, const float* __restrict__ W,
    const float* __restrict__ bias, float* __restrict__ gate,
    float* __restrict__ idx_out) {
  __shared__ float lg[HE];
  const int tok = blockIdx.x;
  const int tid = threadIdx.x;
  const int oid = tid >> 3, sub = tid & 7;
  const float* x = X + (size_t)tok * DDIM;
  const float* w = W + (size_t)oid * DDIM;
  float s = 0.0f;
  for (int i = sub; i < DDIM; i += 8) s += x[i] * w[i];
  s += __shfl_xor(s, 1, 32);
  s += __shfl_xor(s, 2, 32);
  s += __shfl_xor(s, 4, 32);
  if (sub == 0) lg[oid] = s;
  __syncthreads();
  if (tid < HDIM) {
    const int h = tid;
    float sv[EDIM], key[EDIM];
#pragma unroll
    for (int e = 0; e < EDIM; ++e) {
      sv[e]  = 1.0f / (1.0f + __expf(-lg[h * EDIM + e]));
      key[e] = sv[e] + bias[e];
    }
    int i0 = 0;
#pragma unroll
    for (int e = 1; e < EDIM; ++e) if (key[e] > key[i0]) i0 = e;
    int i1 = (i0 == 0) ? 1 : 0;
#pragma unroll
    for (int e = 0; e < EDIM; ++e)
      if (e != i0 && key[e] > key[i1]) i1 = e;
#pragma unroll
    for (int e = 0; e < EDIM; ++e)
      gate[((size_t)tok * HDIM + h) * EDIM + e] =
          (e == i0 || e == i1) ? sv[e] : 0.0f;
    idx_out[((size_t)tok * HDIM + h) * KSEL + 0] = (float)i0;
    idx_out[((size_t)tok * HDIM + h) * KSEL + 1] = (float)i1;
  }
}

// ---------------------------------------------------------------------------
// Elementwise helpers
// ---------------------------------------------------------------------------
__global__ void cast_f32_bf16(const float* __restrict__ in,
                              bf16_t* __restrict__ out, int n) {
  int i = blockIdx.x * blockDim.x + threadIdx.x;
  if (i < n) out[i] = (bf16_t)in[i];
}

// out[k*N + n] = in[n*K + k]  (transpose + cast)
__global__ void packT_f32_bf16(const float* __restrict__ in,
                               bf16_t* __restrict__ out, int Nn, int Kk) {
  int o = blockIdx.x * blockDim.x + threadIdx.x;
  if (o >= Nn * Kk) return;
  int k = o / Nn, n = o % Nn;
  out[o] = (bf16_t)in[(size_t)n * Kk + k];
}

// v_w [HE][D][DH] -> vwT [D][HE*DH]
__global__ void pack_vw(const float* __restrict__ in,
                        bf16_t* __restrict__ out, int n) {
  int o = blockIdx.x * blockDim.x + threadIdx.x;
  if (o >= n) return;
  int d   = o / HEDH;
  int rem = o % HEDH;
  int he  = rem / DHDIM;
  int dh  = rem % DHDIM;
  out[o] = (bf16_t)in[((size_t)he * DDIM + d) * DHDIM + dh];
}

// v_bf[b,h,s,dh] = sum_e gate[b,s,h,e] * all_v[tok, (h*E+e)*DH + dh]
__global__ void combine_v(const bf16_t* __restrict__ allv,
                          const float* __restrict__ gate,
                          bf16_t* __restrict__ vout, int n) {
  int o = blockIdx.x * blockDim.x + threadIdx.x;
  if (o >= n) return;
  int dh = o % DHDIM;
  int s  = (o / DHDIM) % SDIM;
  int h  = (o / (DHDIM * SDIM)) % HDIM;
  int b  = o / (DHDIM * SDIM * HDIM);
  int tok = b * SDIM + s;
  float acc = 0.0f;
#pragma unroll
  for (int e = 0; e < EDIM; ++e)
    acc += gate[((size_t)tok * HDIM + h) * EDIM + e] *
           (float)allv[(size_t)tok * HEDH + (h * EDIM + e) * DHDIM + dh];
  vout[o] = (bf16_t)acc;
}

// scaled[tok, he*DH+dh] = attn_out[tok, h*DH+dh] * o_gate[tok, he]
__global__ void scale_o(const float* __restrict__ attn,
                        const float* __restrict__ gate,
                        bf16_t* __restrict__ out, int n) {
  int o = blockIdx.x * blockDim.x + threadIdx.x;
  if (o >= n) return;
  int dh  = o % DHDIM;
  int he  = (o / DHDIM) % HE;
  int tok = o / HEDH;
  int h   = he / EDIM;
  out[o] = (bf16_t)(attn[(size_t)tok * HDH + h * DHDIM + dh] *
                    gate[(size_t)tok * HE + he]);
}

// ---------------------------------------------------------------------------
// Launcher
// ---------------------------------------------------------------------------
extern "C" void kernel_launch(void* const* d_in, const int* in_sizes, int n_in,
                              void* d_out, int out_size, void* d_ws,
                              size_t ws_size, hipStream_t stream) {
  const float* q_src   = (const float*)d_in[0];
  const float* k_src   = (const float*)d_in[1];
  const float* v_src   = (const float*)d_in[2];
  const float* wq      = (const float*)d_in[3];
  const float* wk      = (const float*)d_in[4];
  const float* sel_v_w = (const float*)d_in[5];
  const float* sel_o_w = (const float*)d_in[6];
  const float* v_w     = (const float*)d_in[7];
  const float* o_w     = (const float*)d_in[8];
  const float* bias_v  = (const float*)d_in[9];
  const float* bias_o  = (const float*)d_in[10];
  // d_in[11..13]: attn_mask (causal tril -> implemented directly),
  //               position_ids, skip_mask (identity) -- unused.
  (void)in_sizes; (void)n_in; (void)out_size; (void)ws_size;

  float* res   = (float*)d_out;                       // [BS, D]
  float* v_idx = res + (size_t)BS * DDIM;             // [BS, H, K] as float
  float* o_idx = v_idx + (size_t)BS * HDIM * KSEL;

  // --- workspace carve ---
  char* p = (char*)d_ws;
  auto alloc = [&](size_t bytes) {
    void* r = (void*)p;
    p += (bytes + 255) & ~(size_t)255;
    return r;
  };
  bf16_t* qsrc_b = (bf16_t*)alloc((size_t)BS * DDIM * 2);
  bf16_t* ksrc_b = (bf16_t*)alloc((size_t)BS * DDIM * 2);
  bf16_t* vsrc_b = (bf16_t*)alloc((size_t)BS * DDIM * 2);
  bf16_t* wqT    = (bf16_t*)alloc((size_t)DDIM * HDH * 2);
  bf16_t* wkT    = (bf16_t*)alloc((size_t)DDIM * HDH * 2);
  bf16_t* vwT    = (bf16_t*)alloc((size_t)DDIM * HEDH * 2);
  bf16_t* owB    = (bf16_t*)alloc((size_t)HEDH * DDIM * 2);
  bf16_t* q_b    = (bf16_t*)alloc((size_t)BS * HDH * 2);
  bf16_t* k_b    = (bf16_t*)alloc((size_t)BS * HDH * 2);
  bf16_t* allv   = (bf16_t*)alloc((size_t)BS * HEDH * 2);
  bf16_t* v_b    = (bf16_t*)alloc((size_t)BS * HDH * 2);      // [B,H,S,DH]
  float*  vgate  = (float*)alloc((size_t)BS * HE * 4);
  float*  ogate  = (float*)alloc((size_t)BS * HE * 4);
  float*  attn_o = (float*)alloc((size_t)BS * HDH * 4);       // [B,S,H,DH]
  bf16_t* scaled = (bf16_t*)alloc((size_t)BS * HEDH * 2);

  const float ssq = powf((float)DHDIM, -0.25f);  // sqrt(1/sqrt(DH))
  const int TB = 256;
  auto blocks = [&](size_t n) { return (unsigned)((n + TB - 1) / TB); };

  // 1) casts / weight packing
  cast_f32_bf16<<<blocks((size_t)BS * DDIM), TB, 0, stream>>>(q_src, qsrc_b, BS * DDIM);
  cast_f32_bf16<<<blocks((size_t)BS * DDIM), TB, 0, stream>>>(k_src, ksrc_b, BS * DDIM);
  cast_f32_bf16<<<blocks((size_t)BS * DDIM), TB, 0, stream>>>(v_src, vsrc_b, BS * DDIM);
  packT_f32_bf16<<<blocks((size_t)HDH * DDIM), TB, 0, stream>>>(wq, wqT, HDH, DDIM);
  packT_f32_bf16<<<blocks((size_t)HDH * DDIM), TB, 0, stream>>>(wk, wkT, HDH, DDIM);
  pack_vw<<<blocks((size_t)DDIM * HEDH), TB, 0, stream>>>(v_w, vwT, DDIM * HEDH);
  cast_f32_bf16<<<blocks((size_t)HEDH * DDIM), TB, 0, stream>>>(o_w, owB, HEDH * DDIM);

  // 2) Q/K projections (ssq folded in) + all-expert V projection
  gemm_bf16<true><<<dim3(HDH / 64, BS / 128), 256, 0, stream>>>(
      qsrc_b, wqT, q_b, BS, HDH, DDIM, ssq);
  gemm_bf16<true><<<dim3(HDH / 64, BS / 128), 256, 0, stream>>>(
      ksrc_b, wkT, k_b, BS, HDH, DDIM, ssq);
  gemm_bf16<true><<<dim3(HEDH / 64, BS / 128), 256, 0, stream>>>(
      vsrc_b, vwT, allv, BS, HEDH, DDIM, 1.0f);

  // 3) gating (V experts on k_src, O experts on q_src); idx into d_out tail
  gate_kernel<<<BS, 256, 0, stream>>>(k_src, sel_v_w, bias_v, vgate, v_idx);
  gate_kernel<<<BS, 256, 0, stream>>>(q_src, sel_o_w, bias_o, ogate, o_idx);

  // 4) gate-combine V -> [B,H,S,DH] bf16
  combine_v<<<blocks((size_t)BS * HDH), TB, 0, stream>>>(
      allv, vgate, v_b, BS * HDH);

  // 5) causal flash attention -> attn_o [B,S,H,DH] f32
  flash_attn<<<dim3(SDIM / 64, BDIM * HDIM), 128, 0, stream>>>(
      q_b, k_b, v_b, attn_o);

  // 6) O-expert scaling + final projection GEMM -> res
  scale_o<<<blocks((size_t)BS * HEDH), TB, 0, stream>>>(
      attn_o, ogate, scaled, BS * HEDH);
  gemm_bf16<false><<<dim3(DDIM / 64, BS / 128), 256, 0, stream>>>(
      scaled, owB, res, BS, DDIM, HEDH, 1.0f);
}
